// CustomTransformerEncoderMoELayer_4097398800715
// MI455X (gfx1250) — compile-verified, matrix operand order checked by reading, not run on hardware
//
#include <hip/hip_runtime.h>
#include <hip/hip_bf16.h>

// ---------------------------------------------------------------------------
// CDNA5 / gfx1250 transformer encoder layer with hash-routed MoE.
// All GEMMs run through v_wmma_f32_16x16x32_bf16 (fp32 accumulate).
// f32->bf16 uses native conversion (v_cvt_pk_bf16_f32 path via __bf16).
// ---------------------------------------------------------------------------

typedef __attribute__((ext_vector_type(16))) __bf16 v16bf;
typedef __attribute__((ext_vector_type(2)))  __bf16 v2bf;
typedef __attribute__((ext_vector_type(8)))  float  v8f;

union Frag {
    v16bf    v;
    unsigned u[8];
};

// native f32 -> bf16 (backend selects packed/scalar bf16 cvt ops)
static __device__ __forceinline__ unsigned pack_bf2(float a, float b) {
    v2bf p;
    p.x = (__bf16)a;
    p.y = (__bf16)b;
    return __builtin_bit_cast(unsigned, p);
}
static __device__ __forceinline__ unsigned short bf16_1(float f) {
    return __builtin_bit_cast(unsigned short, (__bf16)f);
}

#define D_MODEL   1024
#define T_SEQ     512
#define N_TOK     2048
#define D_FF      4096
#define BM        128     // rows per block (= blockDim.x; 1 A-row per thread)
#define BN        64      // cols per block
#define BK        32      // k per stage
// 4 waves / block; each wave computes a 32x64 strip (8 WMMAs per K-stage).

// ---------------------------------------------------------------------------
// Generic tiled bf16-WMMA GEMM, branch-free via templates.
//   MODEA: 0 = dense f32 rows (row index = m)
//          1 = gathered f32 rows (token = list[slot])
//          2 = gathered bf16 rows (packed u32 pairs)
//   EPI:   0 = C = acc + bias            (f32 store)
//          1 = C = relu(acc + bias)      (bf16 store)
//          2 = C = resid + acc + bias    (f32 store)
//          3 = C = acc + bias            (bf16 store, token-major)
//          4 = C = acc + bias            (bf16 store, head-transposed
//                                         [b][h][d][t] via packed b128)
// When cntp != nullptr, B/bias are offset per expert (blockIdx.z) and rows
// with slot >= cnt are masked.
// ---------------------------------------------------------------------------
template <int MODEA, int EPI>
__global__ __launch_bounds__(BM)
void gemm_bf16(const float* __restrict__ Af, const unsigned* __restrict__ Ap,
               const float* __restrict__ Bw, const float* __restrict__ bias,
               const float* __restrict__ resid,
               float* __restrict__ Cf, unsigned short* __restrict__ Cbf,
               const int* __restrict__ rowlist, const int* __restrict__ cntp,
               int M, int N, int K, int ldA)
{
    __shared__ unsigned Alds[BM * 16];
    __shared__ unsigned Blds[BN * 16];
    __shared__ int      toks[BM];

    const int tid  = threadIdx.x;
    const int lane = tid & 31;
    const int wid  = tid >> 5;
    const int half = lane >> 4;      // 0 or 1
    const int lr   = lane & 15;

    const int n0 = blockIdx.x * BN;
    const int m0 = blockIdx.y * BM;

    const int* list = nullptr;
    int cnt = M;
    if (cntp) {
        const int e = blockIdx.z;
        cnt  = cntp[e];
        list = rowlist + e * N_TOK;
        Bw   += (size_t)e * K * N;
        bias += (size_t)e * N;
    }
    if (m0 >= cnt) return;

    // per-thread A row (thread t owns tile row t)
    int atok;
    {
        int slot = m0 + tid;
        if (MODEA == 0)        atok = slot;
        else if (slot < cnt)   atok = list[slot];
        else                   atok = list[0];      // clamp; store masked later
        toks[tid] = atok;
    }
    const float4* a4  = (const float4*)(Af + (size_t)atok * ldA);
    const uint4*  a4u = (const uint4*)(Ap + (size_t)atok * (ldA >> 1));
    __syncthreads();

    v8f acc[2][4] = {};

    const int nkt = K / BK;
    for (int kt = 0; kt < nkt; ++kt) {
        const int kb = kt * BK;
        // prefetch next K-stage (global_prefetch_b8)
        if (kt + 1 < nkt) {
            if (MODEA == 2) __builtin_prefetch((const void*)(a4u + (kt + 1) * 4), 0, 0);
            else            __builtin_prefetch((const void*)(a4 + (kt + 1) * 8), 0, 0);
            __builtin_prefetch((const void*)(Bw + (size_t)(kb + BK) * N + n0 + (tid & 63)), 0, 0);
        }
        // ---- A tile: one full row per thread, vector loads ---------------
        if (MODEA == 2) {
            #pragma unroll
            for (int j = 0; j < 4; ++j) {
                uint4 tv = a4u[kt * 4 + j];
                Alds[(tid << 4) | ((4 * j + 0 + tid) & 15)] = tv.x;
                Alds[(tid << 4) | ((4 * j + 1 + tid) & 15)] = tv.y;
                Alds[(tid << 4) | ((4 * j + 2 + tid) & 15)] = tv.z;
                Alds[(tid << 4) | ((4 * j + 3 + tid) & 15)] = tv.w;
            }
        } else {
            #pragma unroll
            for (int j = 0; j < 8; ++j) {
                float4 tv = a4[kt * 8 + j];
                Alds[(tid << 4) | ((2 * j + 0 + tid) & 15)] = pack_bf2(tv.x, tv.y);
                Alds[(tid << 4) | ((2 * j + 1 + tid) & 15)] = pack_bf2(tv.z, tv.w);
            }
        }
        // ---- B tile: BK x BN, coalesced in n, K-pairs packed per column --
        #pragma unroll
        for (int j = 0; j < 8; ++j) {
            int idx = tid + (j << 7);
            int n   = idx & 63;
            int cu  = idx >> 6;
            const float* br = Bw + (size_t)(kb + cu * 2) * N + n0 + n;
            Blds[(n << 4) | ((cu + n) & 15)] = pack_bf2(br[0], br[N]);
        }
        __syncthreads();

        // ---- fragments + WMMA --------------------------------------------
        Frag fa[2];
        #pragma unroll
        for (int mt = 0; mt < 2; ++mt) {
            const int arw = (wid << 5) + (mt << 4) + lr;
            #pragma unroll
            for (int i = 0; i < 8; ++i) {
                int cu = ((i >> 2) << 3) + (half << 2) + (i & 3);
                fa[mt].u[i] = Alds[(arw << 4) | ((cu + arw) & 15)];
            }
        }
        #pragma unroll
        for (int nt = 0; nt < 4; ++nt) {
            Frag fb;
            const int bn = (nt << 4) + lr;
            #pragma unroll
            for (int i = 0; i < 8; ++i) {
                int cu = ((i >> 2) << 3) + (half << 2) + (i & 3);
                fb.u[i] = Blds[(bn << 4) | ((cu + bn) & 15)];
            }
            acc[0][nt] = __builtin_amdgcn_wmma_f32_16x16x32_bf16(
                false, fa[0].v, false, fb.v, (short)0, acc[0][nt], false, false);
            acc[1][nt] = __builtin_amdgcn_wmma_f32_16x16x32_bf16(
                false, fa[1].v, false, fb.v, (short)0, acc[1][nt], false, false);
        }
        __syncthreads();
    }

    // ---- epilogue ---------------------------------------------------------
    if (EPI == 4) {
        // head-transposed bf16 store: Cbf laid out [b][h][d][t], t-contiguous.
        // Lane's 8 acc values are 8 consecutive tokens -> one b128 store.
        #pragma unroll
        for (int nt = 0; nt < 4; ++nt) {
            int n = n0 + (nt << 4) + lr;           // model column = h*64 + d
            float bvl = bias[n];
            int hd = n >> 6, d = n & 63;
            #pragma unroll
            for (int mt = 0; mt < 2; ++mt) {
                int tok0 = m0 + (wid << 5) + (mt << 4) + (half << 3);
                int bb = tok0 >> 9, t0 = tok0 & (T_SEQ - 1);
                uint4 pk;
                pk.x = pack_bf2(acc[mt][nt][0] + bvl, acc[mt][nt][1] + bvl);
                pk.y = pack_bf2(acc[mt][nt][2] + bvl, acc[mt][nt][3] + bvl);
                pk.z = pack_bf2(acc[mt][nt][4] + bvl, acc[mt][nt][5] + bvl);
                pk.w = pack_bf2(acc[mt][nt][6] + bvl, acc[mt][nt][7] + bvl);
                *(uint4*)(Cbf + ((size_t)(bb * 16 + hd) * 64 + d) * T_SEQ + t0) = pk;
            }
        }
    } else {
        #pragma unroll
        for (int nt = 0; nt < 4; ++nt) {
            int n  = n0 + (nt << 4) + lr;
            float bvl = bias[n];
            #pragma unroll
            for (int mt = 0; mt < 2; ++mt) {
                #pragma unroll
                for (int r = 0; r < 8; ++r) {
                    int rloc = (wid << 5) + (mt << 4) + r + (half << 3);
                    if (cntp && (m0 + rloc) >= cnt) continue;
                    int tok  = toks[rloc];
                    float val = acc[mt][nt][r] + bvl;
                    if (EPI == 1 || EPI == 3) {
                        if (EPI == 1) val = val > 0.f ? val : 0.f;
                        Cbf[(size_t)tok * N + n] = bf16_1(val);
                    } else {
                        if (EPI == 2) val += resid[(size_t)tok * N + n];
                        Cf[(size_t)tok * N + n] = val;
                    }
                }
            }
        }
    }
}

// ---------------------------------------------------------------------------
// Flash attention: one wave per (batch, head, 16-query tile), 8 waves/block.
// K is bf16 token-major; V is bf16 head-transposed [b][h][d][t] — both feed
// fragments with aligned u32 pair-loads, no conversion in the hot loop.
// Computes S^T = K . Q^T so P-fragment assembly is purely in-lane.
// ---------------------------------------------------------------------------
__global__ __launch_bounds__(256)
void attn_kernel(const float* __restrict__ q,
                 const unsigned short* __restrict__ k,
                 const unsigned short* __restrict__ vt,
                 float* __restrict__ o)
{
    const int gid  = blockIdx.x * 8 + (threadIdx.x >> 5);  // B*H*(T/16) = 2048
    const int qt   = gid & 31;
    const int hh   = (gid >> 5) & 15;
    const int b    = gid >> 9;
    const int lane = threadIdx.x & 31;
    const int half = lane >> 4;
    const int lr   = lane & 15;

    const float*          qbase  = q  + ((size_t)b * T_SEQ) * D_MODEL + hh * 64;
    const unsigned short* kbase  = k  + ((size_t)b * T_SEQ) * D_MODEL + hh * 64;
    const unsigned short* vtbase = vt + ((size_t)(b * 16 + hh) * 64) * T_SEQ;

    // Q as B-operand of S^T (lane = query column); fold in 1/sqrt(64)
    Frag fq[2];
    {
        const float* qr = qbase + (size_t)(qt * 16 + lr) * D_MODEL;
        #pragma unroll
        for (int dh = 0; dh < 2; ++dh)
            #pragma unroll
            for (int i = 0; i < 8; ++i) {
                int d = dh * 32 + ((i >> 2) << 4) + (half << 3) + ((i & 3) << 1);
                fq[dh].u[i] = pack_bf2(qr[d] * 0.125f, qr[d + 1] * 0.125f);
            }
    }

    v8f accO[4] = {};
    float m_run = -1e30f, l_run = 0.f;

    for (int jt = 0; jt < T_SEQ / 32; ++jt) {
        // ---- S^T tiles for 32 keys (bf16 K: aligned u32 pair loads) ------
        v8f s[2] = {};
        #pragma unroll
        for (int t2 = 0; t2 < 2; ++t2) {
            const unsigned* kr =
                (const unsigned*)(kbase + (size_t)(jt * 32 + t2 * 16 + lr) * D_MODEL);
            #pragma unroll
            for (int dh = 0; dh < 2; ++dh) {
                Frag fk;
                #pragma unroll
                for (int i = 0; i < 8; ++i) {
                    int d = dh * 32 + ((i >> 2) << 4) + (half << 3) + ((i & 3) << 1);
                    fk.u[i] = kr[d >> 1];
                }
                s[t2] = __builtin_amdgcn_wmma_f32_16x16x32_bf16(
                    false, fk.v, false, fq[dh].v, (short)0, s[t2], false, false);
            }
        }
        // ---- online softmax (stats keyed by query = lane%16) -------------
        float mloc = -1e30f;
        #pragma unroll
        for (int r = 0; r < 8; ++r) {
            mloc = fmaxf(mloc, s[0][r]);
            mloc = fmaxf(mloc, s[1][r]);
        }
        mloc = fmaxf(mloc, __shfl_xor(mloc, 16, 32));
        float m_new = fmaxf(m_run, mloc);
        float p0[8], p1[8], lad = 0.f;
        #pragma unroll
        for (int r = 0; r < 8; ++r) {
            p0[r] = __expf(s[0][r] - m_new);
            p1[r] = __expf(s[1][r] - m_new);
            lad += p0[r] + p1[r];
        }
        lad += __shfl_xor(lad, 16, 32);
        float alpha = __expf(m_run - m_new);
        l_run = l_run * alpha + lad;
        m_run = m_new;

        float arow[8];
        #pragma unroll
        for (int r = 0; r < 8; ++r) arow[r] = __shfl(alpha, r + (half << 3), 32);
        #pragma unroll
        for (int nt = 0; nt < 4; ++nt)
            #pragma unroll
            for (int r = 0; r < 8; ++r) accO[nt][r] *= arow[r];

        // ---- P fragment (in-lane repack) ---------------------------------
        Frag fp;
        #pragma unroll
        for (int i = 0; i < 4; ++i) {
            fp.u[i]     = pack_bf2(p0[2 * i], p0[2 * i + 1]);
            fp.u[i + 4] = pack_bf2(p1[2 * i], p1[2 * i + 1]);
        }
        // ---- O += P @ V (transposed bf16 V: aligned u32 pair loads) ------
        #pragma unroll
        for (int nt = 0; nt < 4; ++nt) {
            Frag fv;
            const unsigned* vr =
                (const unsigned*)(vtbase + (size_t)(nt * 16 + lr) * T_SEQ);
            #pragma unroll
            for (int i = 0; i < 8; ++i) {
                int kk = jt * 32 + ((i >> 2) << 4) + (half << 3) + ((i & 3) << 1);
                fv.u[i] = vr[kk >> 1];
            }
            accO[nt] = __builtin_amdgcn_wmma_f32_16x16x32_bf16(
                false, fp.v, false, fv.v, (short)0, accO[nt], false, false);
        }
    }

    float linv = 1.f / l_run;
    float lrow[8];
    #pragma unroll
    for (int r = 0; r < 8; ++r) lrow[r] = __shfl(linv, r + (half << 3), 32);
    float* obase = o + ((size_t)b * T_SEQ + qt * 16) * D_MODEL + hh * 64;
    #pragma unroll
    for (int nt = 0; nt < 4; ++nt)
        #pragma unroll
        for (int r = 0; r < 8; ++r)
            obase[(size_t)(r + (half << 3)) * D_MODEL + nt * 16 + lr] =
                accO[nt][r] * lrow[r];
}

// ---------------------------------------------------------------------------
// LayerNorm (+ optional second addend, + optional router key / expert id)
// One 256-thread block per token, 4 elements/thread (float4 path).
// ---------------------------------------------------------------------------
__global__ __launch_bounds__(256)
void ln_kernel(const float* __restrict__ a, const float* __restrict__ badd,
               const float* __restrict__ g, const float* __restrict__ be,
               const float* __restrict__ rw, const float* __restrict__ rb,
               float* __restrict__ out, int* __restrict__ eidx)
{
    const int tok = blockIdx.x;
    const int tid = threadIdx.x;
    const float4* ar4 = (const float4*)(a + (size_t)tok * D_MODEL);

    __shared__ float red[8];
    float4 x4 = ar4[tid];
    if (badd) {
        float4 b4 = ((const float4*)(badd + (size_t)tok * D_MODEL))[tid];
        x4.x += b4.x; x4.y += b4.y; x4.z += b4.z; x4.w += b4.w;
    }
    float s = x4.x + x4.y + x4.z + x4.w;
    for (int o2 = 16; o2; o2 >>= 1) s += __shfl_xor(s, o2, 32);
    if ((tid & 31) == 0) red[tid >> 5] = s;
    __syncthreads();
    float tot = 0.f;
    for (int i = 0; i < 8; ++i) tot += red[i];
    float mu = tot * (1.f / D_MODEL);
    __syncthreads();

    float dx = x4.x - mu, dy = x4.y - mu, dz = x4.z - mu, dw = x4.w - mu;
    float s2 = dx * dx + dy * dy + dz * dz + dw * dw;
    for (int o2 = 16; o2; o2 >>= 1) s2 += __shfl_xor(s2, o2, 32);
    if ((tid & 31) == 0) red[tid >> 5] = s2;
    __syncthreads();
    float var = 0.f;
    for (int i = 0; i < 8; ++i) var += red[i];
    float rstd = rsqrtf(var * (1.f / D_MODEL) + 1e-5f);

    float4 g4  = ((const float4*)g)[tid];
    float4 be4 = ((const float4*)be)[tid];
    float4 y4;
    y4.x = dx * rstd * g4.x + be4.x;
    y4.y = dy * rstd * g4.y + be4.y;
    y4.z = dz * rstd * g4.z + be4.z;
    y4.w = dw * rstd * g4.w + be4.w;
    ((float4*)(out + (size_t)tok * D_MODEL))[tid] = y4;

    if (rw) {
        float4 r4 = ((const float4*)rw)[tid];
        float kp = y4.x * r4.x + y4.y * r4.y + y4.z * r4.z + y4.w * r4.w;
        __syncthreads();
        for (int o2 = 16; o2; o2 >>= 1) kp += __shfl_xor(kp, o2, 32);
        if ((tid & 31) == 0) red[tid >> 5] = kp;
        __syncthreads();
        if (tid == 0) {
            float key = rb[0];
            for (int i = 0; i < 8; ++i) key += red[i];
            // jnp.remainder(floor(key), 4): & 3 == positive mod for pow2
            eidx[tok] = ((int)floorf(key)) & 3;
        }
    }
}

__global__ void init_cnt_kernel(int* cnt) {
    if (threadIdx.x < 8) cnt[threadIdx.x] = 0;
}

__global__ __launch_bounds__(256)
void scatter_kernel(const int* __restrict__ eidx, int* cnt, int* list)
{
    int tok = blockIdx.x * blockDim.x + threadIdx.x;
    if (tok < N_TOK) {
        int e = eidx[tok];
        int pos = atomicAdd(&cnt[e], 1);
        list[e * N_TOK + pos] = tok;
    }
}

// ---------------------------------------------------------------------------
extern "C" void kernel_launch(void* const* d_in, const int* in_sizes, int n_in,
                              void* d_out, int out_size, void* d_ws, size_t ws_size,
                              hipStream_t stream)
{
    const float* src  = (const float*)d_in[0];
    const float* Wq   = (const float*)d_in[1];
    const float* bq   = (const float*)d_in[2];
    const float* Wk   = (const float*)d_in[3];
    const float* bk   = (const float*)d_in[4];
    const float* Wv   = (const float*)d_in[5];
    const float* bv   = (const float*)d_in[6];
    const float* Wo   = (const float*)d_in[7];
    const float* bo   = (const float*)d_in[8];
    const float* rw   = (const float*)d_in[9];
    const float* rb   = (const float*)d_in[10];
    const float* W1   = (const float*)d_in[11];
    const float* b1   = (const float*)d_in[12];
    const float* W2   = (const float*)d_in[13];
    const float* b2   = (const float*)d_in[14];
    const float* ln1g = (const float*)d_in[15];
    const float* ln1b = (const float*)d_in[16];
    const float* ln2g = (const float*)d_in[17];
    const float* ln2b = (const float*)d_in[18];

    char* ws = (char*)d_ws;
    const size_t MB = 1024ull * 1024ull;
    float*          qb  = (float*)(ws + 0 * MB);           // 2048x1024 f32
    unsigned short* kbf = (unsigned short*)(ws + 8 * MB);  // 2048x1024 bf16
    unsigned short* vtb = (unsigned short*)(ws + 12 * MB); // [b][h][64][512] bf16
    float*          ab  = (float*)(ws + 16 * MB);          // attention out
    float*          s1  = (float*)(ws + 24 * MB);          // src + attn@Wo
    float*          x1  = (float*)(ws + 32 * MB);          // LN1 output
    unsigned short* hb  = (unsigned short*)(ws + 40 * MB); // 2048x4096 bf16
    float*          ff  = (float*)(ws + 56 * MB);          // expert output
    int*   eidx = (int*)(ws + 64 * MB);
    int*   cnt  = eidx + N_TOK;
    int*   list = cnt + 8;

    dim3 gblk(BM);
    // --- Q (f32), K (bf16), V (bf16 head-transposed) projections ---
    gemm_bf16<0, 0><<<dim3(16, 16, 1), gblk, 0, stream>>>(
        src, nullptr, Wq, bq, nullptr, qb, nullptr, nullptr, nullptr,
        N_TOK, D_MODEL, D_MODEL, D_MODEL);
    gemm_bf16<0, 3><<<dim3(16, 16, 1), gblk, 0, stream>>>(
        src, nullptr, Wk, bk, nullptr, nullptr, kbf, nullptr, nullptr,
        N_TOK, D_MODEL, D_MODEL, D_MODEL);
    gemm_bf16<0, 4><<<dim3(16, 16, 1), gblk, 0, stream>>>(
        src, nullptr, Wv, bv, nullptr, nullptr, vtb, nullptr, nullptr,
        N_TOK, D_MODEL, D_MODEL, D_MODEL);
    // --- attention ---
    attn_kernel<<<dim3(256), dim3(256), 0, stream>>>(qb, kbf, vtb, ab);
    // --- output projection + residual ---
    gemm_bf16<0, 2><<<dim3(16, 16, 1), gblk, 0, stream>>>(
        ab, nullptr, Wo, bo, src, s1, nullptr, nullptr, nullptr,
        N_TOK, D_MODEL, D_MODEL, D_MODEL);
    // --- LN1 + router ---
    ln_kernel<<<dim3(N_TOK), dim3(256), 0, stream>>>(
        s1, nullptr, ln1g, ln1b, rw, rb, x1, eidx);
    init_cnt_kernel<<<dim3(1), dim3(32), 0, stream>>>(cnt);
    scatter_kernel<<<dim3(8), dim3(256), 0, stream>>>(eidx, cnt, list);
    // --- MoE expert GEMMs (gathered rows, masked by per-expert counts) ---
    gemm_bf16<1, 1><<<dim3(D_FF / 64, 16, 4), gblk, 0, stream>>>(
        x1, nullptr, W1, b1, nullptr, nullptr, hb, list, cnt,
        N_TOK, D_FF, D_MODEL, D_MODEL);
    gemm_bf16<2, 0><<<dim3(D_MODEL / 64, 16, 4), gblk, 0, stream>>>(
        nullptr, (const unsigned*)hb, W2, b2, nullptr, ff, nullptr, list, cnt,
        N_TOK, D_MODEL, D_FF, D_FF);
    // --- LN2 (fused residual) -> output ---
    ln_kernel<<<dim3(N_TOK), dim3(256), 0, stream>>>(
        x1, ff, ln2g, ln2b, nullptr, nullptr, (float*)d_out, nullptr);
}
